// SGPBlock_64931315580948
// MI455X (gfx1250) — matrix operand order repaired and checked by reading, not compiled, not run on hardware
//
#include <hip/hip_runtime.h>
#include <hip/hip_bf16.h>
#include <math.h>

// ---------------------------------------------------------------------------
// Problem constants (match reference)
// ---------------------------------------------------------------------------
constexpr int   Bn   = 8;
constexpr int   Cn   = 512;
constexpr int   Tn   = 2048;
constexpr int   HIDn = 4 * Cn;        // 2048
constexpr float EPSf = 1e-5f;
constexpr size_t BCT = (size_t)Bn * Cn * Tn;   // 8388608

typedef __attribute__((ext_vector_type(16))) __bf16 v16bf;
typedef __attribute__((ext_vector_type(8)))  __bf16 v8bf;
typedef __attribute__((ext_vector_type(4)))  __bf16 v4bf;
typedef __attribute__((ext_vector_type(8)))  float  v8f;
typedef __attribute__((ext_vector_type(4)))  unsigned int v4u;
typedef __attribute__((ext_vector_type(8)))  int v8i;
typedef __attribute__((ext_vector_type(4)))  int v4i;

// Tensor Data Mover availability. This toolchain exposes the 6-arg builtin:
//   (uint32x4 g0, int32x8 g1, int32x4 g2, int32x4 g3, int32x8 extra, i32 cpol)
#if defined(__has_builtin)
#if __has_builtin(__builtin_amdgcn_tensor_load_to_lds) && \
    __has_builtin(__builtin_amdgcn_s_wait_tensorcnt)
#define USE_TDM 1
#endif
#endif
#ifndef USE_TDM
#define USE_TDM 0
#endif

// ---------------------------------------------------------------------------
// K0: f32 -> bf16 weight conversion (w1 and w2, both HID*C elements)
// ---------------------------------------------------------------------------
__global__ void k_cvt(const float* __restrict__ w1, const float* __restrict__ w2,
                      __bf16* __restrict__ w1b, __bf16* __restrict__ w2b, int n) {
  int i = blockIdx.x * blockDim.x + threadIdx.x;
  if (i < n) { w1b[i] = (__bf16)w1[i]; w2b[i] = (__bf16)w2[i]; }
}

// ---------------------------------------------------------------------------
// K1: MHA collapses to a constant vector (kv == 0 because UP=7 -> mid=3):
//     local[c] = sum_e aout_w[c,e] * bv[e] + aout_b[c],  bv = ain_b[2C:3C]
// ---------------------------------------------------------------------------
__global__ void k_localvec(const float* __restrict__ aout_w,
                           const float* __restrict__ aout_b,
                           const float* __restrict__ ain_b,
                           float* __restrict__ localv) {
  int c = blockIdx.x * blockDim.x + threadIdx.x;
  if (c >= Cn) return;
  const float* bv = ain_b + 2 * Cn;
  float s = aout_b[c];
  for (int e = 0; e < Cn; ++e) s += aout_w[(size_t)c * Cn + e] * bv[e];
  localv[c] = s;
}

// ---------------------------------------------------------------------------
// K2: channel LayerNorm (reduce over C per (b,t) column). lanes map to t.
// ---------------------------------------------------------------------------
__global__ __launch_bounds__(256) void k_ln(const float* __restrict__ x,
                                            const float* __restrict__ lnw,
                                            const float* __restrict__ lnb,
                                            float* __restrict__ out) {
  int t = blockIdx.x * 256 + threadIdx.x;
  int b = blockIdx.y;
  const float* xp = x + (size_t)b * Cn * Tn + t;
  float s = 0.f, ss = 0.f;
  for (int c = 0; c < Cn; ++c) { float v = xp[(size_t)c * Tn]; s += v; ss += v * v; }
  float mu = s * (1.f / Cn);
  float var = ss * (1.f / Cn) - mu * mu;
  float r = rsqrtf(var + EPSf);
  float* op = out + (size_t)b * Cn * Tn + t;
  for (int c = 0; c < Cn; ++c) {
    float v = (xp[(size_t)c * Tn] - mu) * r;
    op[(size_t)c * Tn] = v * lnw[c] + lnb[c];
  }
}

// ---------------------------------------------------------------------------
// K3: per-(b,c) mean over T -> phi = relu(mean*gfc_w + gfc_b)
// ---------------------------------------------------------------------------
__global__ __launch_bounds__(256) void k_phi(const float* __restrict__ out,
                                             const float* __restrict__ gfcw,
                                             const float* __restrict__ gfcb,
                                             float* __restrict__ phi) {
  int c = blockIdx.x, b = blockIdx.y, tid = threadIdx.x;
  const float* p = out + ((size_t)b * Cn + c) * Tn;
  float s = 0.f;
  for (int i = tid; i < Tn; i += 256) s += p[i];
  __shared__ float red[256];
  red[tid] = s; __syncthreads();
  for (int w = 128; w > 0; w >>= 1) { if (tid < w) red[tid] += red[tid + w]; __syncthreads(); }
  if (tid == 0) {
    float m = red[0] * (1.f / Tn);
    float v = m * gfcw[c] + gfcb[c];
    phi[b * Cn + c] = v > 0.f ? v : 0.f;
  }
}

// ---------------------------------------------------------------------------
// K4: depthwise convs (k=3 and k=7, 'same' zero pad) over LN output
// ---------------------------------------------------------------------------
__global__ __launch_bounds__(256) void k_dwconv(const float* __restrict__ out,
                                                const float* __restrict__ w3,
                                                const float* __restrict__ b3,
                                                const float* __restrict__ w7,
                                                const float* __restrict__ b7,
                                                float* __restrict__ convw,
                                                float* __restrict__ convkw) {
  size_t idx = (size_t)blockIdx.x * 256 + threadIdx.x;
  if (idx >= BCT) return;
  int t = (int)(idx % Tn);
  int c = (int)((idx / Tn) % Cn);
  const float* p = out + (idx - t);
  float v[7];
#pragma unroll
  for (int d = 0; d < 7; ++d) {
    int tt = t + d - 3;
    v[d] = (tt >= 0 && tt < Tn) ? p[tt] : 0.f;
  }
  float a = v[2] * w3[c * 3 + 0] + v[3] * w3[c * 3 + 1] + v[4] * w3[c * 3 + 2] + b3[c];
  float k = b7[c];
#pragma unroll
  for (int d = 0; d < 7; ++d) k += v[d] * w7[c * 7 + d];
  convw[idx] = a;
  convkw[idx] = k;
}

// ---------------------------------------------------------------------------
// K5: gating GEMV: h = relu(g1_w @ [convw;convkw] + g1_b); beta = sigmoid(g2@h+g2_b)
// ---------------------------------------------------------------------------
__global__ __launch_bounds__(256) void k_gate(const float* __restrict__ convw,
                                              const float* __restrict__ convkw,
                                              const float* __restrict__ g1w,
                                              const float* __restrict__ g1b,
                                              const float* __restrict__ g2w,
                                              const float* __restrict__ g2b,
                                              float* __restrict__ beta) {
  int t = blockIdx.x * 256 + threadIdx.x;
  int b = blockIdx.y;
  float h[32];
#pragma unroll
  for (int o = 0; o < 32; ++o) h[o] = g1b[o];
  const float* cwp = convw + (size_t)b * Cn * Tn + t;
  const float* ckp = convkw + (size_t)b * Cn * Tn + t;
  for (int c = 0; c < Cn; ++c) {
    float a = cwp[(size_t)c * Tn];
    float k = ckp[(size_t)c * Tn];
#pragma unroll
    for (int o = 0; o < 32; ++o)
      h[o] += g1w[o * 1024 + c] * a + g1w[o * 1024 + 512 + c] * k;
  }
  float z = g2b[0];
#pragma unroll
  for (int o = 0; o < 32; ++o) z += g2w[o] * (h[o] > 0.f ? h[o] : 0.f);
  beta[b * Tn + t] = 1.f / (1.f + expf(-z));
}

// ---------------------------------------------------------------------------
// K6: combine: out1 = x*mask + local[c] + out + convw*B + (1-B)*convkw + fc*phi
// ---------------------------------------------------------------------------
__global__ __launch_bounds__(256) void k_combine(const float* __restrict__ x,
                                                 const unsigned char* __restrict__ mask,
                                                 const float* __restrict__ out,
                                                 const float* __restrict__ convw,
                                                 const float* __restrict__ convkw,
                                                 const float* __restrict__ beta,
                                                 const float* __restrict__ phi,
                                                 const float* __restrict__ fcw,
                                                 const float* __restrict__ fcb,
                                                 const float* __restrict__ localv,
                                                 float* __restrict__ out1) {
  size_t idx = (size_t)blockIdx.x * 256 + threadIdx.x;
  if (idx >= BCT) return;
  int t = (int)(idx % Tn);
  int c = (int)((idx / Tn) % Cn);
  int b = (int)(idx / ((size_t)Cn * Tn));
  float bt = beta[b * Tn + t];
  float o = out[idx];
  float fc = o * fcw[c] + fcb[c];
  float val = localv[c] + o + convw[idx] * bt + (1.f - bt) * convkw[idx] + fc * phi[b * Cn + c];
  float xm = mask[b * Tn + t] ? x[idx] : 0.f;
  out1[idx] = xm + val;
}

// ---------------------------------------------------------------------------
// K7: GroupNorm(16) stats per (b,group)
// ---------------------------------------------------------------------------
__global__ __launch_bounds__(256) void k_gnstats(const float* __restrict__ out1,
                                                 float* __restrict__ gmean,
                                                 float* __restrict__ grstd) {
  int g = blockIdx.x, b = blockIdx.y, tid = threadIdx.x;
  const float* p = out1 + ((size_t)b * Cn + g * 32) * Tn;
  const int n = 32 * Tn;
  float s = 0.f, ss = 0.f;
  for (int i = tid; i < n; i += 256) { float v = p[i]; s += v; ss += v * v; }
  __shared__ float r1[256], r2[256];
  r1[tid] = s; r2[tid] = ss; __syncthreads();
  for (int w = 128; w > 0; w >>= 1) {
    if (tid < w) { r1[tid] += r1[tid + w]; r2[tid] += r2[tid + w]; }
    __syncthreads();
  }
  if (tid == 0) {
    float m = r1[0] / n;
    float var = r2[0] / n - m * m;
    gmean[b * 16 + g] = m;
    grstd[b * 16 + g] = rsqrtf(var + EPSf);
  }
}

// ---------------------------------------------------------------------------
// K8: normalize + affine, emit bf16 GEMM input
// ---------------------------------------------------------------------------
__global__ __launch_bounds__(256) void k_gnorm(const float* __restrict__ out1,
                                               const float* __restrict__ gmean,
                                               const float* __restrict__ grstd,
                                               const float* __restrict__ gnw,
                                               const float* __restrict__ gnb,
                                               __bf16* __restrict__ g) {
  size_t idx = (size_t)blockIdx.x * 256 + threadIdx.x;
  if (idx >= BCT) return;
  int c = (int)((idx / Tn) % Cn);
  int b = (int)(idx / ((size_t)Cn * Tn));
  int gi = b * 16 + (c >> 5);
  float v = (out1[idx] - gmean[gi]) * grstd[gi] * gnw[c] + gnb[c];
  g[idx] = (__bf16)v;
}

// ---------------------------------------------------------------------------
// WMMA GEMM: per batch b, Y = epi(A[M,K] @ X[b][K,N] + bias[M] (+ res[b]))
// Block tile 128x128, 8 waves (4 in M x 2 in N), wave tile 32x64.
// A tile (128x32 bf16) is moved by the Tensor Data Mover straight into LDS
// (one tensor_load_to_lds per K-step, issued by wave 0, TENSORcnt-tracked);
// TDM pad_enable inserts 4 DWORDs per 16-DWORD row -> LDS row stride of 40
// halfs, identical to the manual layout, so fragment ds_load_b128s and the
// bank-conflict-free strides are unchanged. B tile is loaded by all 8 waves
// and transposed on store (this overlaps with the TDM transfer).
// EPI==0: bias + exact GELU -> bf16.   EPI==1: bias + residual -> f32.
// ---------------------------------------------------------------------------
template <int EPI>
__global__ __launch_bounds__(256) void k_gemm(const __bf16* __restrict__ A,
                                              const __bf16* __restrict__ X,
                                              const float* __restrict__ bias,
                                              const float* __restrict__ res,
                                              void* __restrict__ Y,
                                              int M, int N, int K) {
  const int tid = threadIdx.x;
  const int n0 = blockIdx.x * 128;
  const int m0 = blockIdx.y * 128;
  const int b  = blockIdx.z;
  const __bf16* Xb = X + (size_t)b * K * N;

  __shared__ __bf16 As[128 * 40];
  __shared__ __bf16 Bs[128 * 40];

  const int lane = tid & 31, wave = tid >> 5;
  const int wm = wave >> 1, wn = wave & 1;        // 4x2 wave grid
  const int m16 = lane & 15, hi = lane >> 4;

#if USE_TDM
  const unsigned ldsA = (unsigned)(unsigned long long)(As);  // LDS byte offset
#endif

  v8f acc[2][4];
#pragma unroll
  for (int mi = 0; mi < 2; ++mi)
#pragma unroll
    for (int ni = 0; ni < 4; ++ni) acc[mi][ni] = (v8f)(0.f);

  for (int k0 = 0; k0 < K; k0 += 32) {
    __syncthreads();
#if USE_TDM
    // ---- A tile via Tensor Data Mover (wave 0 only; EXEC-independent DMA) --
    if (wave == 0) {
      unsigned long long ga =
          (unsigned long long)(const void*)&A[(size_t)m0 * K + k0];
      v4u g0;
      g0[0] = 1u;                                         // count=1, load
      g0[1] = ldsA;                                       // lds_addr
      g0[2] = (unsigned)ga;                               // global_addr[31:0]
      g0[3] = (unsigned)((ga >> 32) & 0x01FFFFFFu)        // global_addr[56:32]
              | (2u << 30);                               // type=2 (image)
      v8i g1;
      g1[0] = (int)((1u << 16)          // data_size=1 -> 2 bytes
                    | (1u << 20)        // pad_enable
                    | (3u << 22)        // pad_interval: 16 DWORDs (one row)
                    | (3u << 25));      // pad_amount: 4 DWORDs (8 halfs)
      g1[1] = (int)(((unsigned)K & 0xFFFFu) << 16);       // tensor_dim0 lo
      g1[2] = (int)((((unsigned)K >> 16) & 0xFFFFu)       // tensor_dim0 hi
                    | (128u << 16));                      // tensor_dim1 = 128
      g1[3] = (int)(32u << 16);                           // tile_dim0 = 32
      g1[4] = (int)128u;                                  // tile_dim1 = 128
      g1[5] = (int)K;                                     // dim0_stride lo
      g1[6] = 0;                                          // dim0_stride hi
      g1[7] = 0;
      v4i z4 = {0, 0, 0, 0};                              // 2-D tensor
      v8i z8 = {0, 0, 0, 0, 0, 0, 0, 0};
      __builtin_amdgcn_tensor_load_to_lds(g0, g1, z4, z4, z8, 0);
    }
#else
    // ---- A tile 128x32 (row-major, 16B segments), manual path ----
#pragma unroll
    for (int p = 0; p < 2; ++p) {
      int seg = tid + p * 256;
      int r = seg >> 2, cs = (seg & 3) * 8;
      *(v8bf*)&As[r * 40 + cs] = *(const v8bf*)&A[(size_t)(m0 + r) * K + k0 + cs];
    }
#endif
    // ---- B tile 32x128, transposed to [n][k] (overlaps the TDM transfer) --
#pragma unroll
    for (int p = 0; p < 4; ++p) {
      int seg = tid + p * 256;
      int k = seg >> 5, ns = (seg & 31) * 4;
      v4bf v = *(const v4bf*)&Xb[(size_t)(k0 + k) * N + n0 + ns];
#pragma unroll
      for (int j = 0; j < 4; ++j) Bs[(ns + j) * 40 + k] = v[j];
    }
    // prefetch next K-slab into GL2 (global_prefetch_b8)
    if (k0 + 32 < K) {
      __builtin_prefetch(&A[(size_t)(m0 + (tid >> 1)) * K + k0 + 32], 0, 1);
      __builtin_prefetch(&Xb[(size_t)(k0 + 32 + (tid >> 3)) * N + n0 + (tid & 7) * 16], 0, 1);
    }
#if USE_TDM
    if (wave == 0) __builtin_amdgcn_s_wait_tensorcnt(0);
#endif
    __syncthreads();

    // ---- fragments per ISA 7.12.2 16-bit layouts ----
    v16bf afr[2], bfr[4];
#pragma unroll
    for (int mi = 0; mi < 2; ++mi) {
      const __bf16* ap = &As[(wm * 32 + mi * 16 + m16) * 40];
      v8bf lo = *(const v8bf*)(ap + hi * 8);
      v8bf hh = *(const v8bf*)(ap + 16 + hi * 8);
      afr[mi] = __builtin_shufflevector(lo, hh, 0, 1, 2, 3, 4, 5, 6, 7,
                                        8, 9, 10, 11, 12, 13, 14, 15);
    }
#pragma unroll
    for (int ni = 0; ni < 4; ++ni) {
      const __bf16* bp = &Bs[(wn * 64 + ni * 16 + m16) * 40];
      v8bf lo = *(const v8bf*)(bp + hi * 8);
      v8bf hh = *(const v8bf*)(bp + 16 + hi * 8);
      bfr[ni] = __builtin_shufflevector(lo, hh, 0, 1, 2, 3, 4, 5, 6, 7,
                                        8, 9, 10, 11, 12, 13, 14, 15);
    }
#pragma unroll
    for (int mi = 0; mi < 2; ++mi)
#pragma unroll
      for (int ni = 0; ni < 4; ++ni)
        acc[mi][ni] = __builtin_amdgcn_wmma_f32_16x16x32_bf16(
            false, afr[mi], false, bfr[ni], (short)0, acc[mi][ni], false, false);
  }

  // ---- epilogue (C/D layout: VGPR r -> row r + 8*hi, lane&15 -> col) ----
#pragma unroll
  for (int mi = 0; mi < 2; ++mi) {
#pragma unroll
    for (int ni = 0; ni < 4; ++ni) {
      int mbase = m0 + wm * 32 + mi * 16 + hi * 8;
      int ncol = n0 + wn * 64 + ni * 16 + m16;
#pragma unroll
      for (int r = 0; r < 8; ++r) {
        int m = mbase + r;
        size_t oidx = (size_t)b * M * N + (size_t)m * N + ncol;
        float v = acc[mi][ni][r] + bias[m];
        if (EPI == 0) {
          v = 0.5f * v * (1.f + erff(v * 0.70710678118f));   // exact GELU
          ((__bf16*)Y)[oidx] = (__bf16)v;
        } else {
          ((float*)Y)[oidx] = v + res[oidx];
        }
      }
    }
  }
}

// ---------------------------------------------------------------------------
// K9: mask output tail (bool ones -> 1.0f)
// ---------------------------------------------------------------------------
__global__ void k_mask(const unsigned char* __restrict__ mask, float* __restrict__ o) {
  int i = blockIdx.x * 256 + threadIdx.x;
  if (i < Bn * Tn) o[i] = mask[i] ? 1.f : 0.f;
}

// ---------------------------------------------------------------------------
extern "C" void kernel_launch(void* const* d_in, const int* in_sizes, int n_in,
                              void* d_out, int out_size, void* d_ws, size_t ws_size,
                              hipStream_t stream) {
  const float* x      = (const float*)d_in[0];
  const unsigned char* mask = (const unsigned char*)d_in[1];
  const float* ln_w   = (const float*)d_in[2];
  const float* ln_b   = (const float*)d_in[3];
  const float* fc_w   = (const float*)d_in[4];
  const float* fc_b   = (const float*)d_in[5];
  const float* convw_w  = (const float*)d_in[6];
  const float* convw_b  = (const float*)d_in[7];
  const float* convkw_w = (const float*)d_in[8];
  const float* convkw_b = (const float*)d_in[9];
  const float* gfc_w  = (const float*)d_in[10];
  const float* gfc_b  = (const float*)d_in[11];
  const float* g1_w   = (const float*)d_in[12];
  const float* g1_b   = (const float*)d_in[13];
  const float* g2_w   = (const float*)d_in[14];
  const float* g2_b   = (const float*)d_in[15];
  const float* ain_b  = (const float*)d_in[17];
  const float* aout_w = (const float*)d_in[18];
  const float* aout_b = (const float*)d_in[19];
  const float* gn_w   = (const float*)d_in[20];
  const float* gn_b   = (const float*)d_in[21];
  const float* w1     = (const float*)d_in[22];
  const float* b1     = (const float*)d_in[23];
  const float* w2     = (const float*)d_in[24];
  const float* b2     = (const float*)d_in[25];

  char* wsb = (char*)d_ws;
  float*  out_f  = (float*)(wsb);                 // [B,C,T] LN output (32MB)
  float*  convw  = (float*)(wsb + BCT * 4);       // 32MB
  float*  convkw = (float*)(wsb + BCT * 8);       // 32MB
  float*  out1   = (float*)(wsb + BCT * 12);      // 32MB
  __bf16* hm     = (__bf16*)(wsb + BCT * 4);      // reuses convw+convkw (64MB bf16)
  __bf16* gbuf   = (__bf16*)(wsb);                // reuses out_f (16MB bf16)
  size_t  off    = BCT * 16;
  __bf16* w1b    = (__bf16*)(wsb + off);  off += (size_t)HIDn * Cn * 2;
  __bf16* w2b    = (__bf16*)(wsb + off);  off += (size_t)Cn * HIDn * 2;
  float*  phi    = (float*)(wsb + off);   off += (size_t)Bn * Cn * 4;
  float*  beta   = (float*)(wsb + off);   off += (size_t)Bn * Tn * 4;
  float*  gmean  = (float*)(wsb + off);   off += (size_t)Bn * 16 * 4;
  float*  grstd  = (float*)(wsb + off);   off += (size_t)Bn * 16 * 4;
  float*  localv = (float*)(wsb + off);   off += (size_t)Cn * 4;

  float* y = (float*)d_out;

  k_cvt<<<(HIDn * Cn + 255) / 256, 256, 0, stream>>>(w1, w2, w1b, w2b, HIDn * Cn);
  k_localvec<<<(Cn + 255) / 256, 256, 0, stream>>>(aout_w, aout_b, ain_b, localv);
  k_ln<<<dim3(Tn / 256, Bn), 256, 0, stream>>>(x, ln_w, ln_b, out_f);
  k_phi<<<dim3(Cn, Bn), 256, 0, stream>>>(out_f, gfc_w, gfc_b, phi);
  k_dwconv<<<(unsigned)(BCT / 256), 256, 0, stream>>>(out_f, convw_w, convw_b,
                                                      convkw_w, convkw_b, convw, convkw);
  k_gate<<<dim3(Tn / 256, Bn), 256, 0, stream>>>(convw, convkw, g1_w, g1_b, g2_w, g2_b, beta);
  k_combine<<<(unsigned)(BCT / 256), 256, 0, stream>>>(x, mask, out_f, convw, convkw,
                                                       beta, phi, fc_w, fc_b, localv, out1);
  k_gnstats<<<dim3(16, Bn), 256, 0, stream>>>(out1, gmean, grstd);
  k_gnorm<<<(unsigned)(BCT / 256), 256, 0, stream>>>(out1, gmean, grstd, gn_w, gn_b, gbuf);
  // GEMM1: hm = gelu(w1 @ g + b1)   M=2048, N=2048, K=512
  k_gemm<0><<<dim3(Tn / 128, HIDn / 128, Bn), 256, 0, stream>>>(
      w1b, gbuf, b1, nullptr, (void*)hm, HIDn, Tn, Cn);
  // GEMM2: out = out1 + w2 @ hm + b2   M=512, N=2048, K=2048
  k_gemm<1><<<dim3(Tn / 128, Cn / 128, Bn), 256, 0, stream>>>(
      w2b, hm, b2, out1, (void*)y, Cn, Tn, HIDn);
  k_mask<<<(Bn * Tn + 255) / 256, 256, 0, stream>>>(mask, y + BCT);
}
